// GRUD_ODE_36498632081581
// MI455X (gfx1250) — matrix operand order repaired and checked by reading, not compiled, not run on hardware
//
#include <hip/hip_runtime.h>
#include <hip/hip_bf16.h>
#include <stdint.h>

// Problem dims
#define T_STEPS 8192
#define I_SIZE  256
#define H_SIZE  1024
#define K_DIM   512           // 2*I : [mask | imputed x]
#define N_DIM   3072          // 3*H : gates r,z,u concatenated
#define NUM_RWG 64            // workgroups in the recurrent persistent kernel
#define ROWS_PER_WG 16        // H / NUM_RWG
#define WSTRIDE 1040          // padded LDS row stride (bank-conflict avoidance)

typedef __attribute__((ext_vector_type(16))) _Float16 v16h;
typedef __attribute__((ext_vector_type(8)))  _Float16 v8h;
typedef __attribute__((ext_vector_type(8)))  float    v8f;
typedef __attribute__((ext_vector_type(4)))  unsigned int u32x4;
typedef __attribute__((ext_vector_type(8)))  int i32x8;
typedef __attribute__((ext_vector_type(4)))  int i32x4;

// ---------------- workspace layout (bytes) ----------------
#define WS_A16   0u                                   // 8192*512*2   = 8 MB   (f16 A)
#define WS_B16   (WS_A16 + (size_t)T_STEPS*K_DIM*2)   // 3072*512*2   = 3 MB   (f16 B^T)
#define WS_BIAS  (WS_B16 + (size_t)N_DIM*K_DIM*2)     // 3072*4
#define WS_G     (WS_BIAS + (size_t)N_DIM*4)          // 8192*3072*4  = 96 MB
#define WS_H     (WS_G + (size_t)T_STEPS*N_DIM*4)     // 1024*4
#define WS_RH    (WS_H + (size_t)H_SIZE*4)            // 1024*4
#define WS_CTRL  (WS_RH + (size_t)H_SIZE*4)           // cnt, gen, t_eff

__device__ __forceinline__ float sigmoidf_dev(float x) {
    return 1.0f / (1.0f + __expf(-x));
}

// ------------------------------------------------------------------
// TDM: issue a 2D tile load (tile 32 x 128 rows of f16, row stride 512
// elements) from global memory into LDS. Descriptor per CDNA5 ISA §8.
// Must be reached by exactly one wave; EXEC is ignored by the TDM op.
// ------------------------------------------------------------------
__device__ __forceinline__ void tdm_load_tile_f16(uint32_t lds_addr,
                                                  const _Float16* gptr,
                                                  uint32_t tensor_rows)
{
    const uint64_t ga = (uint64_t)(uintptr_t)gptr;
    u32x4 g0;
    g0[0] = 1u;                                         // count=1 (valid D#)
    g0[1] = lds_addr;                                   // lds_addr[31:0] (bytes)
    g0[2] = (uint32_t)(ga & 0xFFFFFFFFu);               // global_addr[31:0]
    g0[3] = (uint32_t)((ga >> 32) & 0x1FFFFFFu)         // global_addr[56:32]
          | (2u << 30);                                 // type = 2 ("image")
    i32x8 g1;
    g1[0] = (int)(1u << 16);          // data_size=1 (2 bytes); mask/pad/iter = 0
    g1[1] = (int)(512u << 16);        // tensor_dim0[15:0] = 512 (@bit48)
    g1[2] = (int)(tensor_rows << 16); // tensor_dim0[31:16]=0 ; tensor_dim1[15:0]
    g1[3] = (int)(32u << 16);         // tensor_dim1[31:16]=0 ; tile_dim0 = 32
    g1[4] = 128;                      // tile_dim1 = 128 ; tile_dim2 = 0
    g1[5] = 512;                      // tensor_dim0_stride[31:0] = 512
    g1[6] = 0;                        // stride hi / tensor_dim1_stride lo
    g1[7] = 0;
    const i32x4 z4 = {0, 0, 0, 0};    // groups 2/3 unused (2D tensor)
#if defined(__clang_major__) && (__clang_major__ >= 23)
    const i32x8 z8 = {0, 0, 0, 0, 0, 0, 0, 0};
    __builtin_amdgcn_tensor_load_to_lds(g0, g1, z4, z4, z8, 0);
#else
    __builtin_amdgcn_tensor_load_to_lds(g0, g1, z4, z4, 0);
#endif
}

// ------------------------------------------------------------------
// K0: T_eff, state init, imputation scan -> builds f16 A = [Mask|Ximp]
// One block, 256 threads (one thread per input channel).
// ------------------------------------------------------------------
__global__ __launch_bounds__(256)
void scan_prep(const float* __restrict__ X, const float* __restrict__ Mask,
               const float* __restrict__ dt,
               _Float16* __restrict__ A16, float* __restrict__ h_g,
               float* __restrict__ rh_g, uint32_t* __restrict__ ctrl)
{
    const int tid = threadIdx.x;
    if (tid == 0) {
        uint32_t te = T_STEPS;
        for (int t = 0; t < T_STEPS; ++t) {
            if (dt[t] == 0.0f) { te = (uint32_t)t; break; }
        }
        ctrl[0] = 0u;   // barrier count
        ctrl[1] = 0u;   // barrier generation
        ctrl[2] = te;   // effective step count
    }
#pragma unroll
    for (int q = 0; q < 4; ++q) {
        h_g[tid * 4 + q]  = 0.0f;
        rh_g[tid * 4 + q] = 0.0f;
    }
    // GRU-D imputation scan for channel `tid` (independent of h).
    float prex = 0.0f;
    const float* Xi = &X[(size_t)tid * T_STEPS];
    const float* Mi = &Mask[(size_t)tid * T_STEPS];
    for (int t = 0; t < T_STEPS; ++t) {
        const float m = Mi[t];
        const float x = Xi[t];
        const float xt = fmaf(m, x - prex, prex);   // m*x + (1-m)*prex
        A16[(size_t)t * K_DIM + tid]          = (_Float16)m;
        A16[(size_t)t * K_DIM + I_SIZE + tid] = (_Float16)xt;
        prex = xt;
    }
}

// ------------------------------------------------------------------
// K1: pack B^T[3072][512] (f16) = [[W_m*^T];[W_x*^T]] per gate, + bias
// ------------------------------------------------------------------
__global__ __launch_bounds__(256)
void pack_B(const float* __restrict__ Wm0, const float* __restrict__ Wm1,
            const float* __restrict__ Wm2, const float* __restrict__ Wx0,
            const float* __restrict__ Wx1, const float* __restrict__ Wx2,
            const float* __restrict__ bx0, const float* __restrict__ bx1,
            const float* __restrict__ bx2,
            _Float16* __restrict__ Bt, float* __restrict__ bias)
{
    const int idx = blockIdx.x * 256 + threadIdx.x;   // 0 .. 3072*512-1
    const int col = idx >> 9;        // output column (gate*1024 + j)
    const int k   = idx & 511;       // reduction index
    const int g   = col >> 10;       // gate
    const int jj  = col & 1023;      // hidden row
    const float* Wm = (g == 0) ? Wm0 : ((g == 1) ? Wm1 : Wm2);
    const float* Wx = (g == 0) ? Wx0 : ((g == 1) ? Wx1 : Wx2);
    const float v = (k < I_SIZE) ? Wm[jj * I_SIZE + k]
                                 : Wx[jj * I_SIZE + (k - I_SIZE)];
    Bt[(size_t)col * K_DIM + k] = (_Float16)v;
    if (k == 0) {
        const float* bx = (g == 0) ? bx0 : ((g == 1) ? bx1 : bx2);
        bias[col] = bx[jj];
    }
}

// ------------------------------------------------------------------
// K2: WMMA GEMM  G[8192][3072] = A[8192][512] * B^T + bias
// Block: 128x128 tile, 8 waves (wave32), wave tile 32x64 = 2x4 WMMA tiles.
// Tiles staged by the Tensor Data Mover (tensor_load_to_lds), double
// buffered, synchronized on TENSORcnt + workgroup barrier.
// ------------------------------------------------------------------
__global__ __launch_bounds__(256)
void gemm_gates(const _Float16* __restrict__ A, const _Float16* __restrict__ Bt,
                const float* __restrict__ bias, float* __restrict__ G)
{
    __shared__ __align__(16) _Float16 As[2][128 * 32];
    __shared__ __align__(16) _Float16 Bs[2][128 * 32];

    const int tid  = threadIdx.x;
    const int m0   = blockIdx.y * 128;
    const int n0   = blockIdx.x * 128;
    const int wid  = tid >> 5;
    const int lane = tid & 31;
    const int wm   = (wid & 3) * 32;   // 4 waves along M
    const int wn   = (wid >> 2) * 64;  // 2 waves along N
    const int lrow  = lane & 15;
    const int khalf = lane >> 4;

    v8f acc[2][4];
#pragma unroll
    for (int i = 0; i < 2; ++i)
#pragma unroll
        for (int j = 0; j < 4; ++j) acc[i][j] = (v8f)0.0f;

    // Prologue: TDM-load k-chunk 0 into buffer 0 (wave 0 issues; EXEC ignored).
    if (tid == 0) {
        tdm_load_tile_f16((uint32_t)(uintptr_t)&As[0][0],
                          &A[(size_t)m0 * K_DIM], T_STEPS);
        tdm_load_tile_f16((uint32_t)(uintptr_t)&Bs[0][0],
                          &Bt[(size_t)n0 * K_DIM], N_DIM);
    }

    for (int it = 0; it < K_DIM / 32; ++it) {
        const int cb = it & 1;
        if (tid == 0) {
            if (it < K_DIM / 32 - 1) {
                const int kn = (it + 1) * 32;
                tdm_load_tile_f16((uint32_t)(uintptr_t)&As[cb ^ 1][0],
                                  &A[(size_t)m0 * K_DIM + kn], T_STEPS);
                tdm_load_tile_f16((uint32_t)(uintptr_t)&Bs[cb ^ 1][0],
                                  &Bt[(size_t)n0 * K_DIM + kn], N_DIM);
                // in-order TENSORcnt: <=2 outstanding => current buffer done
                __builtin_amdgcn_s_wait_tensorcnt((short)2);
            } else {
                __builtin_amdgcn_s_wait_tensorcnt((short)0);
            }
        }
        __syncthreads();   // publish TDM-written LDS tile to all waves

        // A fragment: lane holds row M=lane%16; halves [0..7]=K(khalf*8..+7),
        // [8..15]=K(16+khalf*8..+7) per ISA 16-bit A 16x32 layout.
        v16h afr[2];
#pragma unroll
        for (int mt = 0; mt < 2; ++mt) {
            const _Float16* p = &As[cb][(wm + mt * 16 + lrow) * 32];
            v8h lo = *reinterpret_cast<const v8h*>(p + khalf * 8);
            v8h hi = *reinterpret_cast<const v8h*>(p + 16 + khalf * 8);
            afr[mt] = __builtin_shufflevector(lo, hi, 0, 1, 2, 3, 4, 5, 6, 7,
                                              8, 9, 10, 11, 12, 13, 14, 15);
        }
        // B fragment: lane holds col N=lane%16; halves [0..7]=K(khalf*8..+7),
        // [8..15]=K(16+khalf*8..+7), matching A.
        v16h bfr[4];
#pragma unroll
        for (int nt = 0; nt < 4; ++nt) {
            const _Float16* p = &Bs[cb][(wn + nt * 16 + lrow) * 32];
            v8h lo = *reinterpret_cast<const v8h*>(p + khalf * 8);
            v8h hi = *reinterpret_cast<const v8h*>(p + 16 + khalf * 8);
            bfr[nt] = __builtin_shufflevector(lo, hi, 0, 1, 2, 3, 4, 5, 6, 7,
                                              8, 9, 10, 11, 12, 13, 14, 15);
        }
#pragma unroll
        for (int mt = 0; mt < 2; ++mt)
#pragma unroll
            for (int nt = 0; nt < 4; ++nt)
                acc[mt][nt] = __builtin_amdgcn_wmma_f32_16x16x32_f16(
                    false, afr[mt], false, bfr[nt], (short)0, acc[mt][nt],
                    false, false);
        __syncthreads();   // all waves done reading before buffer reuse
    }

    // Epilogue: C layout -> row = khalf*8 + e, col = lane%16 (per tile).
#pragma unroll
    for (int nt = 0; nt < 4; ++nt) {
        const int col = n0 + wn + nt * 16 + lrow;
        const float b = bias[col];
#pragma unroll
        for (int mt = 0; mt < 2; ++mt) {
#pragma unroll
            for (int e = 0; e < 8; ++e) {
                const int row = m0 + wm + mt * 16 + khalf * 8 + e;
                G[(size_t)row * N_DIM + col] = acc[mt][nt][e] + b;
            }
        }
    }
}

// ------------------------------------------------------------------
// Software grid barrier (sense-reversing, agent-scope atomics)
// ------------------------------------------------------------------
__device__ __forceinline__ void grid_barrier(uint32_t* cnt, uint32_t* gen,
                                             uint32_t nwg)
{
    __threadfence();                 // make this thread's agent stores visible
    __syncthreads();
    if (threadIdx.x == 0) {
        const uint32_t g = __hip_atomic_load(gen, __ATOMIC_RELAXED,
                                             __HIP_MEMORY_SCOPE_AGENT);
        const uint32_t a = __hip_atomic_fetch_add(cnt, 1u, __ATOMIC_ACQ_REL,
                                                  __HIP_MEMORY_SCOPE_AGENT);
        if (a == nwg - 1u) {
            __hip_atomic_store(cnt, 0u, __ATOMIC_RELAXED,
                               __HIP_MEMORY_SCOPE_AGENT);
            __hip_atomic_fetch_add(gen, 1u, __ATOMIC_RELEASE,
                                   __HIP_MEMORY_SCOPE_AGENT);
        } else {
            while (__hip_atomic_load(gen, __ATOMIC_ACQUIRE,
                                     __HIP_MEMORY_SCOPE_AGENT) == g)
                __builtin_amdgcn_s_sleep(2);
        }
    }
    __syncthreads();
}

// ------------------------------------------------------------------
// K3: persistent recurrent kernel. 64 WGs x 256 threads.
// Each WG owns 16 hidden rows; W_hr/W_hz/W_hu slices live in LDS (f32,
// padded stride) for the whole 8192-step scan. Two grid barriers/step.
// Thread map: s = tid&15 (K-slice), r = tid>>4 (owned row).
// ------------------------------------------------------------------
__global__ __launch_bounds__(256)
void grud_recurrent(const float* __restrict__ Whr, const float* __restrict__ Whz,
                    const float* __restrict__ Whu, const float* __restrict__ G,
                    float* __restrict__ h_g, float* __restrict__ rh_g,
                    uint32_t* __restrict__ ctrl)
{
    extern __shared__ float smf[];
    float* Wl  = smf;                              // 3*16*WSTRIDE floats
    float* hs  = smf + 3 * ROWS_PER_WG * WSTRIDE;  // 1024
    float* rhs = hs + H_SIZE;                      // 1024

    const int tid = threadIdx.x;
    const int wg  = blockIdx.x;
    const int s   = tid & 15;
    const int r   = tid >> 4;
    const int base_row = wg * ROWS_PER_WG;

    // Stage this WG's weight rows into LDS (float4, coalesced).
    for (int lin = tid; lin < 3 * ROWS_PER_WG * (H_SIZE / 4); lin += 256) {
        const int g   = lin / (ROWS_PER_WG * (H_SIZE / 4));
        const int rem = lin % (ROWS_PER_WG * (H_SIZE / 4));
        const int row = rem / (H_SIZE / 4);
        const int k4  = rem % (H_SIZE / 4);
        const float* W = (g == 0) ? Whr : ((g == 1) ? Whz : Whu);
        const float4 v = *reinterpret_cast<const float4*>(
            &W[(size_t)(base_row + row) * H_SIZE + k4 * 4]);
        *reinterpret_cast<float4*>(&Wl[(g * ROWS_PER_WG + row) * WSTRIDE + k4 * 4]) = v;
    }
    const uint32_t T_eff = __hip_atomic_load(&ctrl[2], __ATOMIC_RELAXED,
                                             __HIP_MEMORY_SCOPE_AGENT);
    __syncthreads();

    const int j = base_row + r;
    const float* Wr_row = &Wl[(0 * ROWS_PER_WG + r) * WSTRIDE];
    const float* Wz_row = &Wl[(1 * ROWS_PER_WG + r) * WSTRIDE];
    const float* Wu_row = &Wl[(2 * ROWS_PER_WG + r) * WSTRIDE];

    float zv = 0.0f, hj = 0.0f;   // live only on lanes with s==0

    for (uint32_t t = 0; t < T_eff; ++t) {
        // ---- Phase A: r and z gates -------------------------------
#pragma unroll
        for (int q = 0; q < 4; ++q) {
            const int idx = tid * 4 + q;
            hs[idx] = __hip_atomic_load(&h_g[idx], __ATOMIC_RELAXED,
                                        __HIP_MEMORY_SCOPE_AGENT);
        }
        __syncthreads();

        float ar = 0.0f, az = 0.0f;
#pragma unroll 8
        for (int i = 0; i < 64; ++i) {
            const int k = s + (i << 4);
            const float hv = hs[k];
            ar = fmaf(hv, Wr_row[k], ar);
            az = fmaf(hv, Wz_row[k], az);
        }
#pragma unroll
        for (int m = 8; m >= 1; m >>= 1) {
            ar += __shfl_xor(ar, m, 16);
            az += __shfl_xor(az, m, 16);
        }
        if (s == 0) {
            const float* Gt = &G[(size_t)t * N_DIM];
            const float rv = sigmoidf_dev(Gt[j] + ar);
            zv = sigmoidf_dev(Gt[H_SIZE + j] + az);
            hj = hs[j];
            __hip_atomic_store(&rh_g[j], rv * hj, __ATOMIC_RELAXED,
                               __HIP_MEMORY_SCOPE_AGENT);
        }
        grid_barrier(&ctrl[0], &ctrl[1], NUM_RWG);

        // ---- Phase B: candidate u, hidden update ------------------
#pragma unroll
        for (int q = 0; q < 4; ++q) {
            const int idx = tid * 4 + q;
            rhs[idx] = __hip_atomic_load(&rh_g[idx], __ATOMIC_RELAXED,
                                         __HIP_MEMORY_SCOPE_AGENT);
        }
        __syncthreads();

        float au = 0.0f;
#pragma unroll 8
        for (int i = 0; i < 64; ++i) {
            const int k = s + (i << 4);
            au = fmaf(rhs[k], Wu_row[k], au);
        }
#pragma unroll
        for (int m = 8; m >= 1; m >>= 1) au += __shfl_xor(au, m, 16);

        if (s == 0) {
            const float uv = tanhf(G[(size_t)t * N_DIM + 2 * H_SIZE + j] + au);
            const float hn = (1.0f - zv) * hj + zv * uv;
            __hip_atomic_store(&h_g[j], hn, __ATOMIC_RELAXED,
                               __HIP_MEMORY_SCOPE_AGENT);
        }
        grid_barrier(&ctrl[0], &ctrl[1], NUM_RWG);
    }
}

// ------------------------------------------------------------------
// K4: out = sigmoid(h . W_out + b_out)   (O_SIZE == 1)
// ------------------------------------------------------------------
__global__ __launch_bounds__(256)
void out_kernel(const float* __restrict__ h_g, const float* __restrict__ W_out,
                const float* __restrict__ b_out, float* __restrict__ out)
{
    __shared__ float red[256];
    const int tid = threadIdx.x;
    float acc = 0.0f;
    for (int k = tid; k < H_SIZE; k += 256)
        acc += __hip_atomic_load(&h_g[k], __ATOMIC_RELAXED,
                                 __HIP_MEMORY_SCOPE_AGENT) * W_out[k];
    red[tid] = acc;
    __syncthreads();
    for (int w = 128; w > 0; w >>= 1) {
        if (tid < w) red[tid] += red[tid + w];
        __syncthreads();
    }
    if (tid == 0) out[0] = sigmoidf_dev(red[0] + b_out[0]);
}

// ------------------------------------------------------------------
extern "C" void kernel_launch(void* const* d_in, const int* in_sizes, int n_in,
                              void* d_out, int out_size, void* d_ws, size_t ws_size,
                              hipStream_t stream)
{
    (void)in_sizes; (void)n_in; (void)out_size; (void)ws_size;

    const float* X    = (const float*)d_in[0];
    const float* Mask = (const float*)d_in[1];
    // d_in[2] = Delta (unused by reference)
    const float* dt   = (const float*)d_in[3];
    // d_in[4..7] = w_dg_x, w_dg_h, b_dg_x, b_dg_h (unused by reference)
    const float* W_xh = (const float*)d_in[8];
    const float* b_xh = (const float*)d_in[9];
    const float* W_xz = (const float*)d_in[10];
    const float* b_xz = (const float*)d_in[11];
    const float* W_xr = (const float*)d_in[12];
    const float* b_xr = (const float*)d_in[13];
    const float* W_hu = (const float*)d_in[14];
    const float* W_hz = (const float*)d_in[15];
    const float* W_hr = (const float*)d_in[16];
    const float* W_mu = (const float*)d_in[17];
    const float* W_mz = (const float*)d_in[18];
    const float* W_mr = (const float*)d_in[19];
    const float* W_out = (const float*)d_in[20];
    const float* b_out = (const float*)d_in[21];

    char* ws = (char*)d_ws;
    _Float16* A16  = (_Float16*)(ws + WS_A16);
    _Float16* B16  = (_Float16*)(ws + WS_B16);
    float*    bias = (float*)(ws + WS_BIAS);
    float*    G    = (float*)(ws + WS_G);
    float*    h_g  = (float*)(ws + WS_H);
    float*    rh_g = (float*)(ws + WS_RH);
    uint32_t* ctrl = (uint32_t*)(ws + WS_CTRL);

    // K0: T_eff + state init + imputation scan -> A16
    scan_prep<<<1, 256, 0, stream>>>(X, Mask, dt, A16, h_g, rh_g, ctrl);

    // K1: pack f16 B^T (gate order r,z,u) + fused bias
    pack_B<<<(N_DIM * K_DIM) / 256, 256, 0, stream>>>(
        W_mr, W_mz, W_mu, W_xr, W_xz, W_xh, b_xr, b_xz, b_xh, B16, bias);

    // K2: WMMA GEMM (TDM-staged) -> G[t][gate*1024 + j]
    dim3 ggrid(N_DIM / 128, T_STEPS / 128);
    gemm_gates<<<ggrid, 256, 0, stream>>>(A16, B16, bias, G);

    // K3: persistent recurrent scan (dynamic LDS: weights + h + rh)
    const size_t lds_bytes = (size_t)(3 * ROWS_PER_WG * WSTRIDE + 2 * H_SIZE) * 4;
    grud_recurrent<<<NUM_RWG, 256, lds_bytes, stream>>>(
        W_hr, W_hz, W_hu, G, h_g, rh_g, ctrl);

    // K4: output head
    out_kernel<<<1, 256, 0, stream>>>(h_g, W_out, b_out, (float*)d_out);
}
